// RTF_83502754169193
// MI455X (gfx1250) — matrix operand order repaired and checked by reading, not compiled, not run on hardware
//
#include <hip/hip_runtime.h>
#include <cstdint>
#include <cstddef>

// Problem constants (from reference): u (B,L,D) fp32, ab (2D,N) fp32, h0 (D) fp32
#define B_   4
#define L_   4096
#define D_   1024
#define N_   64
// Reversed, zero-padded kernel image per channel:
//   kpad[j]: j in [0,32) -> 0 (negative lags), kpad[32+t] = k[t], tail zeros
//   krev[j] = kpad[KTOT-1-j];  KTOT = 4352
#define KTOT 4352
#define ULDS_B 4192   // per-batch padded u row in LDS: 96 zero prefix + 4096
#define P1B  192      // phase-1 block size (hist fits in 48KB LDS)

typedef __attribute__((ext_vector_type(16))) __bf16  v16bf;
typedef __attribute__((ext_vector_type(8)))  float   v8f;
typedef __attribute__((ext_vector_type(4)))  unsigned u32x4;
typedef __attribute__((ext_vector_type(4)))  int      i32x4;
typedef __attribute__((ext_vector_type(8)))  int      i32x8;

__device__ __forceinline__ unsigned short f2bfu(float f) {
  union { float f; unsigned u; } x; x.f = f;
  unsigned r = x.u + 0x7FFFu + ((x.u >> 16) & 1u);   // RNE
  return (unsigned short)(r >> 16);
}
__device__ __forceinline__ __bf16 us2bf(unsigned short u) {
  union { unsigned short u; __bf16 b; } x; x.u = u; return x.b;
}
// Generic LDS pointer -> wave-relative LDS byte offset (aperture: addr[31:0])
__device__ __forceinline__ unsigned lds_off(const void* p) {
  return (unsigned)(unsigned long long)p;
}

// ---------------------------------------------------------------------------
// Phase 0: transpose/pack u (B,L,D) fp32 -> ut[d][b][t] bf16 (coalesced both ways)
// ---------------------------------------------------------------------------
__global__ void rtf_pack_u(const float* __restrict__ u, unsigned short* __restrict__ ut) {
  __shared__ float tile[32][33];
  const int x = threadIdx.x, y = threadIdx.y;
  const int d0 = blockIdx.x * 32, t0 = blockIdx.y * 32, b = blockIdx.z;
  #pragma unroll
  for (int yy = y; yy < 32; yy += 8)
    tile[yy][x] = u[((size_t)b * L_ + (size_t)(t0 + yy)) * D_ + (d0 + x)];
  __syncthreads();
  #pragma unroll
  for (int yy = y; yy < 32; yy += 8)
    ut[(size_t)(d0 + yy) * (4 * L_) + (size_t)b * L_ + (t0 + x)] = f2bfu(tile[x][yy]);
}

// ---------------------------------------------------------------------------
// Phase 1: per-channel IIR recurrence -> reversed+padded bf16 kernel image
//   h_t = b_t - sum_{n=1..64} a_n h_{t-n};  k[0]=h0, k[t]=h_t
// ---------------------------------------------------------------------------
__global__ void __launch_bounds__(P1B)
rtf_build_k(const float* __restrict__ ab, const float* __restrict__ h0,
            unsigned short* __restrict__ krev) {
  __shared__ float hist[64 * P1B];            // ring buffer, one column per thread
  const int tid = threadIdx.x;
  const int d = blockIdx.x * P1B + tid;
  if (d >= D_) return;
  const float* ag = ab + (size_t)d * N_;            // denominator taps a_1..a_64
  const float* bg = ab + (size_t)(D_ + d) * N_;     // numerator taps b_1..b_64
  float ac[64];
  #pragma unroll
  for (int n = 0; n < 64; ++n) ac[n] = ag[n];
  #pragma unroll
  for (int s = 0; s < 64; ++s) hist[s * P1B + tid] = 0.f;  // h_{t<0} = 0

  unsigned short* kr = krev + (size_t)d * KTOT;
  for (int j = 0; j < 224; ++j) kr[j] = 0;          // lags >= 4096 (reversed front)
  for (int j = 4320; j < KTOT; ++j) kr[j] = 0;      // negative lags (reversed tail)

  const float h0v = h0[d];
  for (int t = 0; t < L_; ++t) {
    float s = 0.f;
    if (t >= 1 && t <= 64) s = bg[t - 1];
    #pragma unroll
    for (int n = 0; n < 64; ++n)
      s -= ac[n] * hist[(((t - 1 - n) & 63)) * P1B + tid];
    hist[(t & 63) * P1B + tid] = s;
    kr[KTOT - 33 - t] = f2bfu(t == 0 ? h0v : s);
  }
}

// ---------------------------------------------------------------------------
// Phase 2: depthwise causal convolution as blocked-Toeplitz bf16 WMMA.
// One workgroup = one channel d, 8 waves. Column packing per WMMA:
//   n = 4c+b -> output block T+2c, batch b; input 32-block S+c
//   block-lag lam = 16T - 32S constant over all 16 columns -> shared A tile
// LDS fill: u-slice via global_load_async_to_lds_b128 (ASYNCcnt), kernel image
// via TDM tensor_load_to_lds (TENSORcnt) issued once by wave 0.
// ---------------------------------------------------------------------------
__global__ void __launch_bounds__(256)
rtf_conv_wmma(const unsigned short* __restrict__ ut,
              const unsigned short* __restrict__ krev,
              float* __restrict__ y) {
  __shared__ __attribute__((aligned(32))) unsigned short u_lds[B_ * ULDS_B];  // 33536 B
  __shared__ __attribute__((aligned(32))) unsigned short k_lds[KTOT];         //  8704 B

  const int tid = threadIdx.x;
  const int d = blockIdx.x;

  // ---- async global->LDS fill of the packed u slice (16B per lane per issue) ----
  {
    const unsigned ubase = lds_off(&u_lds[0]);
    const unsigned long long usrc =
        (unsigned long long)(const void*)(ut + (size_t)d * (4 * L_));
    #pragma unroll
    for (int it = 0; it < 8; ++it) {                       // 2048 16B chunks / 256 thr
      const int q  = tid + 256 * it;
      const int s0 = q * 8;
      const int bb = s0 >> 12;                             // /4096
      const int t  = s0 & (L_ - 1);
      const unsigned dst = ubase + (unsigned)((bb * ULDS_B + 96 + t) * 2);
      const unsigned long long ga = usrc + (unsigned long long)(q * 16);
      asm volatile("global_load_async_to_lds_b128 %0, %1, off"
                   :: "v"(dst), "v"(ga) : "memory");
    }
    const uint4 z = make_uint4(0u, 0u, 0u, 0u);
    if (tid < 48) {                                        // 4 * 96 zero prefix
      const int bb = tid / 12, off = (tid % 12) * 8;
      *(uint4*)(&u_lds[bb * ULDS_B + off]) = z;
    }
  }

  // ---- TDM: one descriptor copies the 8704B reversed-kernel row into LDS ----
  if (tid < 32) {                                          // wave 0 only (EXEC ignored)
    const unsigned kbase = lds_off(&k_lds[0]);
    const unsigned long long ga =
        (unsigned long long)(const void*)(krev + (size_t)d * KTOT);
    // D# group 0: count=1, lds_addr, 57b global addr, type=2
    u32x4 g0 = { 1u, kbase, (unsigned)ga,
                 (unsigned)((ga >> 32) & 0x01FFFFFFull) | 0x80000000u };
    // D# group 1: data_size=2B (code 1); tensor_dim0=tile_dim0=stride0=KTOT;
    // tensor_dim1=tile_dim1=1; dim1_stride=KTOT; no pad/iterate/barrier/mask.
    i32x8 g1 = { 0x00010000,
                 (int)((KTOT & 0xFFFF) << 16),
                 (int)(((unsigned)KTOT >> 16) | (1u << 16)),
                 (int)((KTOT & 0xFFFF) << 16),
                 1,
                 KTOT,
                 (int)((KTOT & 0xFFFF) << 16),
                 0 };
    i32x4 gz = { 0, 0, 0, 0 };
#if defined(__clang_major__) && (__clang_major__ >= 23)
    i32x8 gz8 = { 0, 0, 0, 0, 0, 0, 0, 0 };
    __builtin_amdgcn_tensor_load_to_lds(g0, g1, gz, gz, gz8, 0);
#else
    __builtin_amdgcn_tensor_load_to_lds(g0, g1, gz, gz, 0);
#endif
  }

  __builtin_amdgcn_s_wait_tensorcnt(0);
  asm volatile("s_wait_asynccnt 0x0" ::: "memory");
  __syncthreads();

  // ---- wave/lane decomposition ----
  const int lane = tid & 31;
  const int w    = tid >> 5;          // 8 waves
  const int hgrp = lane >> 4;         // A/B lane-group (K half)
  const int Mrow = lane & 15;         // A row
  const int ncol = lane & 15;         // D/B column
  const int ccol = ncol >> 2;         // output-block offset within group
  const int bcol = ncol & 3;          // batch
  const int p  = w & 1;               // T parity for this wave
  const int wh = w >> 1;

  v8f acc[8];
  #pragma unroll
  for (int j = 0; j < 8; ++j) acc[j] = (v8f){0.f,0.f,0.f,0.f,0.f,0.f,0.f,0.f};

  const int mlim = 4 * wh + 115;      // last m with any valid group (S_7 >= -3)
  for (int m = 0; m <= mlim; ++m) {
    const int lam   = 16 * p + 32 * m;
    const int abase = KTOT - 33 - lam - Mrow;
    v16bf A;
    #pragma unroll
    for (int e = 0; e < 16; ++e) {
      const int kap = 16 * hgrp + e;                     // A K-layout contiguous (k = 16h + e)
      A[e] = us2bf(k_lds[abase + kap]);
    }
    const int Sbase = 4 * wh - m;
    #pragma unroll
    for (int j = 0; j < 8; ++j) {
      const int S = Sbase + 16 * j;
      if (S >= -3) {                                     // wave-uniform => EXEC stays full
        const int uoff = bcol * ULDS_B + 96 + 32 * (S + ccol) + 16 * hgrp;
        const v16bf Bm = *(const v16bf*)(&u_lds[uoff]);
        acc[j] = __builtin_amdgcn_wmma_f32_16x16x32_bf16(
            false, A, false, Bm, (short)0, acc[j], false, false);
      }
    }
  }

  // ---- D-matrix writeback: row M = r + 8*hgrp, col n=4c+b ----
  #pragma unroll
  for (int j = 0; j < 8; ++j) {
    const int T  = 8 * wh + p + 32 * j;
    const int t0 = 16 * (T + 2 * ccol) + 8 * hgrp;
    float* op = y + (size_t)bcol * L_ * D_ + (size_t)t0 * D_ + d;
    #pragma unroll
    for (int r = 0; r < 8; ++r) op[(size_t)r * D_] = acc[j][r];
  }
}

// ---------------------------------------------------------------------------
extern "C" void kernel_launch(void* const* d_in, const int* in_sizes, int n_in,
                              void* d_out, int out_size, void* d_ws, size_t ws_size,
                              hipStream_t stream) {
  (void)in_sizes; (void)n_in; (void)out_size; (void)ws_size;
  const float* u  = (const float*)d_in[0];
  const float* ab = (const float*)d_in[1];
  const float* h0 = (const float*)d_in[2];
  float* y = (float*)d_out;

  unsigned short* ut   = (unsigned short*)d_ws;                                    // 32 MiB
  unsigned short* krev = (unsigned short*)((char*)d_ws + (size_t)D_ * 4 * L_ * 2); // 8.5 MiB

  dim3 g0(D_ / 32, L_ / 32, B_), b0(32, 8, 1);
  rtf_pack_u<<<g0, b0, 0, stream>>>(u, ut);
  rtf_build_k<<<(D_ + P1B - 1) / P1B, P1B, 0, stream>>>(ab, h0, krev);
  rtf_conv_wmma<<<D_, 256, 0, stream>>>(ut, krev, y);
}
